// encoder_63324997812512
// MI455X (gfx1250) — matrix operand order repaired and checked by reading, not compiled, not run on hardware
//
#include <hip/hip_runtime.h>
#include <hip/hip_bf16.h>

typedef __attribute__((ext_vector_type(16))) _Float16 v16h;
typedef __attribute__((ext_vector_type(8)))  _Float16 v8h;
typedef __attribute__((ext_vector_type(8)))  float    v8f;

#define NB    8      // batch elements per block
#define AST   136    // ldsA stride (halves): 272B/row -> 4-dword bank shift, conflict-free
#define BST   264    // ldsB stride (halves): 528B/row -> 4-dword bank shift, conflict-free
#define W1ST  264
#define W2ST  136

__device__ __forceinline__ float xshfl(float v, int mask) {
  int l = (int)(threadIdx.x & 31);
  return __int_as_float(__builtin_amdgcn_ds_bpermute(((l ^ mask) << 2),
                                                     __float_as_int(v)));
}

__device__ __forceinline__ void cvt_store8(_Float16* dst, const float* src) {
  const float4 f0 = *(const float4*)(src);
  const float4 f1 = *(const float4*)(src + 4);
  v8h h;
  h[0] = (_Float16)f0.x; h[1] = (_Float16)f0.y;
  h[2] = (_Float16)f0.z; h[3] = (_Float16)f0.w;
  h[4] = (_Float16)f1.x; h[5] = (_Float16)f1.y;
  h[6] = (_Float16)f1.z; h[7] = (_Float16)f1.w;
  *(v8h*)dst = h;
}

__device__ __forceinline__ float leaky(float x) { return x > 0.f ? x : 0.2f * x; }

__global__ __launch_bounds__(128)
void encoder_kernel(const int* __restrict__ nodes_u,
                    const int* __restrict__ un_table,
                    const int* __restrict__ u2l_table,
                    const float* __restrict__ u_emb,
                    const float* __restrict__ u_loc_emb,
                    const float* __restrict__ w1,  const float* __restrict__ b1,
                    const float* __restrict__ w2,  const float* __restrict__ b2,
                    const float* __restrict__ att_w1, const float* __restrict__ att_b1,
                    const float* __restrict__ att_w2, const float* __restrict__ att_b2,
                    const float* __restrict__ cg, const float* __restrict__ cb,
                    const float* __restrict__ cm, const float* __restrict__ cv,
                    const float* __restrict__ sg, const float* __restrict__ sb,
                    const float* __restrict__ sm, const float* __restrict__ sv,
                    float* __restrict__ out, int Bn) {
  __shared__ __align__(32) _Float16 ldsB[128 * BST];   // att_w1 (row n, col k) f16
  __shared__ __align__(32) _Float16 ldsA[64 * AST];    // neighbor features f16 (64x128)
  __shared__ __align__(32) _Float16 ldsW1[64 * W1ST];  // w1 f16
  __shared__ __align__(32) _Float16 ldsW2[64 * W2ST];  // w2 f16
  __shared__ float    s_uembf[128];
  __shared__ _Float16 s_uembh[128];
  __shared__ int   s_inter[64], s_lidx[64];
  __shared__ float s_scores[64], s_attw[64];
  __shared__ float s_comb[256];
  __shared__ float s_attb1[128], s_attw2[128];
  __shared__ float s_b1[64], s_b2[64];
  __shared__ float s_cscale[64], s_cshift[64], s_sscale[64], s_sshift[64];

  const int tid  = threadIdx.x;
  const int lane = tid & 31;
  const int wave = tid >> 5;
  const int colSel  = lane & 15;
  const int hiHalf  = lane >> 4;     // 0: lanes 0-15, 1: lanes 16-31
  const int lo8     = hiHalf * 8;    // A-fragment K sub-offset
  const int bK16    = hiHalf * 16;   // unused; B fragment loads use lo8
  const float attb2 = att_b2[0];

  // ---- stage weights into LDS (once per block) ----
  for (int c = tid; c < 128 * 32; c += 128) {           // att_w1: 128 rows x 256
    int row = c >> 5, col = (c & 31) * 8;
    cvt_store8(&ldsB[row * BST + col], att_w1 + row * 256 + col);
  }
  for (int c = tid; c < 64 * 32; c += 128) {            // w1: 64 x 256
    int row = c >> 5, col = (c & 31) * 8;
    cvt_store8(&ldsW1[row * W1ST + col], w1 + row * 256 + col);
  }
  for (int c = tid; c < 64 * 16; c += 128) {            // w2: 64 x 128
    int row = c >> 4, col = (c & 15) * 8;
    cvt_store8(&ldsW2[row * W2ST + col], w2 + row * 128 + col);
  }
  if (tid < 128) { s_attb1[tid] = att_b1[tid]; s_attw2[tid] = att_w2[tid]; }
  if (tid < 64) {
    s_b1[tid] = b1[tid];  s_b2[tid] = b2[tid];
    float sc = cg[tid] * rsqrtf(cv[tid] + 1e-5f);
    s_cscale[tid] = sc;   s_cshift[tid] = cb[tid] - cm[tid] * sc;
    float ss = sg[tid] * rsqrtf(sv[tid] + 1e-5f);
    s_sscale[tid] = ss;   s_sshift[tid] = sb[tid] - sm[tid] * ss;
  }
  __syncthreads();

  for (int ib = 0; ib < NB; ++ib) {
    const int b = blockIdx.x * NB + ib;
    if (b < Bn) {
      const int ids  = nodes_u[2 * b];
      const int locs = nodes_u[2 * b + 1];

      // ---- gather: indices + user embedding ----
      if (tid < 64) {
        int nb = un_table[(size_t)ids * 64 + tid];
        s_inter[tid] = nb;
        s_lidx[tid]  = u2l_table[nb];
      }
      {
        float v = (tid < 64) ? u_emb[(size_t)ids * 64 + tid]
                             : u_loc_emb[(size_t)locs * 64 + (tid - 64)];
        s_uembf[tid] = v;
        s_uembh[tid] = (_Float16)v;
      }
      __syncthreads();

      // ---- gather: 64 neighbor rows (u_emb | u_loc_emb) -> ldsA f16 ----
      {
        int k = tid >> 1, half = tid & 1;
        const float* src = half ? (u_loc_emb + (size_t)s_lidx[k] * 64)
                                : (u_emb     + (size_t)s_inter[k] * 64);
        _Float16* dst = &ldsA[k * AST + half * 64];
#pragma unroll
        for (int i = 0; i < 64; i += 8) cvt_store8(dst + i, src + i);
      }
      __syncthreads();

      // ---- attention GEMM: h = leaky(att_in @ att_w1^T + b) fused into scores ----
      {
        const int m0 = wave * 16;
        float ps[8];
#pragma unroll
        for (int r = 0; r < 8; ++r) ps[r] = 0.f;

        for (int nt = 0; nt < 8; ++nt) {
          const int n = nt * 16 + colSel;
          const _Float16* bbase = &ldsB[n * BST];
          v8f c = {};
#pragma unroll
          for (int kt = 0; kt < 8; ++kt) {
            const int kk0 = kt * 32;
            const _Float16* arow = (kk0 < 128)
                ? &ldsA[(m0 + colSel) * AST + kk0]
                : &s_uembh[kk0 - 128];            // broadcast user-embedding cols
            v8h alo = *(const v8h*)(arow + lo8);
            v8h ahi = *(const v8h*)(arow + 16 + lo8);
            v16h a = __builtin_shufflevector(alo, ahi,
                        0,1,2,3,4,5,6,7,8,9,10,11,12,13,14,15);
            v8h blo = *(const v8h*)(bbase + kk0 + lo8);
            v8h bhi = *(const v8h*)(bbase + kk0 + 16 + lo8);
            v16h bf = __builtin_shufflevector(blo, bhi,
                        0,1,2,3,4,5,6,7,8,9,10,11,12,13,14,15);
            c = __builtin_amdgcn_wmma_f32_16x16x32_f16(
                    false, a, false, bf, (short)0, c, false, false);
          }
          const float ab1 = s_attb1[n];
          const float aw2 = s_attw2[n];
#pragma unroll
          for (int r = 0; r < 8; ++r) ps[r] += leaky(c[r] + ab1) * aw2;
        }
        // reduce over the 16 lanes of each half (columns of the tile)
#pragma unroll
        for (int mask = 1; mask <= 8; mask <<= 1)
#pragma unroll
          for (int r = 0; r < 8; ++r) ps[r] += xshfl(ps[r], mask);
        if (colSel == 0) {
#pragma unroll
          for (int r = 0; r < 8; ++r)
            s_scores[m0 + hiHalf * 8 + r] = ps[r] + attb2;
        }
      }
      __syncthreads();

      // ---- softmax over 64 scores (wave 0) ----
      if (wave == 0) {
        float s0 = s_scores[lane], s1 = s_scores[lane + 32];
        float mx = fmaxf(s0, s1);
#pragma unroll
        for (int mask = 1; mask <= 16; mask <<= 1) mx = fmaxf(mx, xshfl(mx, mask));
        float e0 = __expf(s0 - mx), e1 = __expf(s1 - mx);
        float sum = e0 + e1;
#pragma unroll
        for (int mask = 1; mask <= 16; mask <<= 1) sum += xshfl(sum, mask);
        float inv = 1.f / sum;
        s_attw[lane] = e0 * inv;
        s_attw[lane + 32] = e1 * inv;
      }
      __syncthreads();

      // ---- weighted neighbor feature + combined vector ----
      {
        float acc = 0.f;
        for (int k = 0; k < 64; ++k)
          acc += s_attw[k] * (float)ldsA[k * AST + tid];
        s_comb[128 + tid] = acc;
        s_comb[tid] = s_uembf[tid];
      }
      __syncthreads();

      // ---- final matvecs + BN + leaky + add ----
      if (tid < 64) {
        const int j = tid;
        float accC = s_b1[j];
        const _Float16* w1r = &ldsW1[j * W1ST];
#pragma unroll 8
        for (int i = 0; i < 256; ++i) accC += s_comb[i] * (float)w1r[i];
        float accS = s_b2[j];
        const _Float16* w2r = &ldsW2[j * W2ST];
#pragma unroll 8
        for (int i = 0; i < 128; ++i) accS += s_uembf[i] * (float)w2r[i];
        float ce = leaky(accC * s_cscale[j] + s_cshift[j]);
        float se = leaky(accS * s_sscale[j] + s_sshift[j]);
        out[(size_t)b * 64 + j] = ce + se;
      }
    }
    __syncthreads();   // protect shared buffers before next element's gather
  }
}

extern "C" void kernel_launch(void* const* d_in, const int* in_sizes, int n_in,
                              void* d_out, int out_size, void* d_ws, size_t ws_size,
                              hipStream_t stream) {
  const int*   nodes_u   = (const int*)d_in[0];
  const int*   un_table  = (const int*)d_in[1];
  const int*   u2l_table = (const int*)d_in[2];
  const float* u_emb     = (const float*)d_in[3];
  const float* u_loc_emb = (const float*)d_in[4];
  const float* w1        = (const float*)d_in[5];
  const float* b1        = (const float*)d_in[6];
  const float* w2        = (const float*)d_in[7];
  const float* b2        = (const float*)d_in[8];
  const float* att_w1    = (const float*)d_in[9];
  const float* att_b1    = (const float*)d_in[10];
  const float* att_w2    = (const float*)d_in[11];
  const float* att_b2    = (const float*)d_in[12];
  const float* cg        = (const float*)d_in[13];
  const float* cb        = (const float*)d_in[14];
  const float* cm        = (const float*)d_in[15];
  const float* cv        = (const float*)d_in[16];
  const float* sg        = (const float*)d_in[17];
  const float* sb        = (const float*)d_in[18];
  const float* sm        = (const float*)d_in[19];
  const float* sv        = (const float*)d_in[20];
  float* out = (float*)d_out;

  const int Bn = in_sizes[0] / 2;
  const int grid = (Bn + NB - 1) / NB;
  encoder_kernel<<<grid, 128, 0, stream>>>(
      nodes_u, un_table, u2l_table, u_emb, u_loc_emb,
      w1, b1, w2, b2, att_w1, att_b1, att_w2, att_b2,
      cg, cb, cm, cv, sg, sb, sm, sv, out, Bn);
}
